// CausalSelfAttention_36292473651775
// MI455X (gfx1250) — matrix operand order repaired and checked by reading, not compiled, not run on hardware
//
#include <hip/hip_runtime.h>

typedef __attribute__((ext_vector_type(16))) __bf16 bf16x16;
typedef __attribute__((ext_vector_type(8)))  __bf16 bf16x8;
typedef __attribute__((ext_vector_type(8)))  float  f32x8;
typedef __attribute__((ext_vector_type(8)))  unsigned short u16x8;

#define WMMA_BF16(a, b, c) \
  __builtin_amdgcn_wmma_f32_16x16x32_bf16(false, (a), false, (b), (short)0, (c), false, false)

__device__ __forceinline__ unsigned short f2bf(float f) {
  unsigned int u = __builtin_bit_cast(unsigned int, f);
  unsigned int r = u + 0x7FFFu + ((u >> 16) & 1u);
  return (unsigned short)(r >> 16);
}

__device__ __forceinline__ bf16x8 ld8(const unsigned short* p) {
  return *reinterpret_cast<const bf16x8*>(p);
}

__device__ __forceinline__ u16x8 gld8(const unsigned short* p) {
  return __builtin_bit_cast(u16x8, *reinterpret_cast<const bf16x8*>(p));
}

__device__ __forceinline__ bf16x16 cat16(bf16x8 lo, bf16x8 hi) {
  bf16x16 r;
#pragma unroll
  for (int i = 0; i < 8; ++i) { r[i] = lo[i]; r[i + 8] = hi[i]; }
  return r;
}

__device__ __forceinline__ f32x8 fzero() {
  f32x8 z;
#pragma unroll
  for (int i = 0; i < 8; ++i) z[i] = 0.0f;
  return z;
}

// Wave-relative LDS byte offset of a __shared__ object (ISA: LDS addr = low 32
// bits of the generic address).
__device__ __forceinline__ unsigned lds_off(const void* p) {
  return (unsigned)(uintptr_t)p;
}

// CDNA5 async global->LDS copy, 16 bytes per lane, tracked by ASYNCcnt.
__device__ __forceinline__ void async_copy_b128(unsigned lds_byte_off, const void* gptr) {
  asm volatile("global_load_async_to_lds_b128 %0, %1, off"
               :: "v"(lds_byte_off), "v"(gptr) : "memory");
}
__device__ __forceinline__ void async_wait0() {
  asm volatile("s_wait_asynccnt 0" ::: "memory");
}

// ---------------------------------------------------------------------------
// fp32 -> bf16 elementwise convert (4 elems / thread)
// ---------------------------------------------------------------------------
__global__ void cvt_f32_bf16(const float* __restrict__ in,
                             unsigned short* __restrict__ out, int n) {
  int i = (blockIdx.x * blockDim.x + threadIdx.x) * 4;
  if (i < n) {
    float4 f = *reinterpret_cast<const float4*>(in + i);
    uint2 o;
    o.x = (unsigned)f2bf(f.x) | ((unsigned)f2bf(f.y) << 16);
    o.y = (unsigned)f2bf(f.z) | ((unsigned)f2bf(f.w) << 16);
    *reinterpret_cast<uint2*>(out + i) = o;
  }
}

// ---------------------------------------------------------------------------
// QKV GEMM: [8192,768]bf16 x [768,2304]bf16 + bias -> q/k/v bf16 [B,H,T,D]
// block tile 128(M) x 64(N); double-buffered LDS: A via async global->LDS,
// W tile global->regs overlapped with WMMA, transposed ds stores after.
// ---------------------------------------------------------------------------
__global__ __launch_bounds__(128) void qkv_gemm(
    const unsigned short* __restrict__ xb,   // [8192][768]
    const unsigned short* __restrict__ wb,   // [768][2304]
    const float* __restrict__ bias,          // [2304]
    unsigned short* __restrict__ qb,
    unsigned short* __restrict__ kb,
    unsigned short* __restrict__ vb) {
  __shared__ unsigned short As[2][128 * 40];  // [m][k], stride 40
  __shared__ unsigned short Bs[2][64 * 40];   // [n][k] (transposed W tile)

  const int tid = threadIdx.x;
  const int wid = tid >> 5, lane = tid & 31;
  const int l15 = lane & 15, h = lane >> 4;
  const int m0 = blockIdx.y * 128;
  const int n0 = blockIdx.x * 64;

  int ar[4], ac[4];
#pragma unroll
  for (int it = 0; it < 4; ++it) { int e = tid * 8 + it * 1024; ar[it] = e >> 5; ac[it] = e & 31; }
  int br[2], bc[2];
#pragma unroll
  for (int it = 0; it < 2; ++it) { int e = tid * 8 + it * 1024; br[it] = e >> 6; bc[it] = e & 63; }

  f32x8 acc[2][4];
#pragma unroll
  for (int i = 0; i < 2; ++i)
#pragma unroll
    for (int j = 0; j < 4; ++j) acc[i][j] = fzero();

  // ---- prologue: stage k-tile 0 into buffer 0
#pragma unroll
  for (int it = 0; it < 4; ++it)
    async_copy_b128(lds_off(&As[0][ar[it] * 40 + ac[it]]),
                    xb + (size_t)(m0 + ar[it]) * 768 + ac[it]);
  {
    u16x8 breg[2];
#pragma unroll
    for (int it = 0; it < 2; ++it)
      breg[it] = gld8(wb + (size_t)br[it] * 2304 + n0 + bc[it]);
#pragma unroll
    for (int it = 0; it < 2; ++it)
#pragma unroll
      for (int j = 0; j < 8; ++j) Bs[0][(bc[it] + j) * 40 + br[it]] = breg[it][j];
  }

  for (int kt = 0; kt < 24; ++kt) {
    const int buf = kt & 1;
    async_wait0();
    __syncthreads();

    const bool more = (kt + 1) < 24;
    u16x8 breg[2];
    if (more) {                                // stage k-tile kt+1 into buf^1
      const int kn = (kt + 1) * 32;
#pragma unroll
      for (int it = 0; it < 4; ++it)
        async_copy_b128(lds_off(&As[buf ^ 1][ar[it] * 40 + ac[it]]),
                        xb + (size_t)(m0 + ar[it]) * 768 + kn + ac[it]);
#pragma unroll
      for (int it = 0; it < 2; ++it)
        breg[it] = gld8(wb + (size_t)(kn + br[it]) * 2304 + n0 + bc[it]);
    }

    // ---- compute on buffer `buf` (overlaps with in-flight loads)
    bf16x16 a[2];
#pragma unroll
    for (int ms = 0; ms < 2; ++ms) {
      const int am = wid * 32 + ms * 16 + l15;
      a[ms] = cat16(ld8(&As[buf][am * 40 + 8 * h]),
                    ld8(&As[buf][am * 40 + 16 + 8 * h]));
    }
#pragma unroll
    for (int nt = 0; nt < 4; ++nt) {
      const int bn = nt * 16 + l15;
      bf16x16 b = cat16(ld8(&Bs[buf][bn * 40 + 8 * h]),
                        ld8(&Bs[buf][bn * 40 + 16 + 8 * h]));
#pragma unroll
      for (int ms = 0; ms < 2; ++ms) acc[ms][nt] = WMMA_BF16(a[ms], b, acc[ms][nt]);
    }

    if (more) {                                // W data arrived during compute
#pragma unroll
      for (int it = 0; it < 2; ++it)
#pragma unroll
        for (int j = 0; j < 8; ++j) Bs[buf ^ 1][(bc[it] + j) * 40 + br[it]] = breg[it][j];
    }
  }

  // Epilogue: which of q/k/v is uniform per block (n0 = blockIdx.x*64, 768%64==0).
  const int which = blockIdx.x / 12;
  unsigned short* dst = (which == 0) ? qb : (which == 1) ? kb : vb;
  const float scl = (which == 0) ? 0.125f : 1.0f;   // fold 1/sqrt(64) into q
  const int cbase = n0 - which * 768;
#pragma unroll
  for (int ms = 0; ms < 2; ++ms) {
#pragma unroll
    for (int nt = 0; nt < 4; ++nt) {
#pragma unroll
      for (int r = 0; r < 8; ++r) {
        int gm = m0 + wid * 32 + ms * 16 + r + 8 * h;
        int cc = cbase + nt * 16 + l15;
        int head = cc >> 6, d = cc & 63;
        int bb = gm >> 10, t = gm & 1023;
        float val = (acc[ms][nt][r] + bias[n0 + nt * 16 + l15]) * scl;
        dst[((size_t)(bb * 12 + head) * 1024 + t) * 64 + d] = f2bf(val);
      }
    }
  }
}

// ---------------------------------------------------------------------------
// Flash attention: one workgroup = (b,h) x 64 query rows; 4 waves x 16 rows.
// Double-buffered K/V tiles: K async global->LDS, V global->regs overlapped
// with WMMA + softmax, transposed ds stores after compute.
// ---------------------------------------------------------------------------
__global__ __launch_bounds__(128) void attn_kernel(
    const unsigned short* __restrict__ qbuf,
    const unsigned short* __restrict__ kbuf,
    const unsigned short* __restrict__ vbuf,
    unsigned short* __restrict__ yb) {
  __shared__ unsigned short Ks[2][32 * 72];    // [key][d], stride 72
  __shared__ unsigned short Vs[2][64 * 40];    // [d][key], stride 40 (transposed)
  __shared__ unsigned short Ps[4 * 16 * 40];   // per-wave P 16x32, stride 40

  const int tid = threadIdx.x;
  const int wid = tid >> 5, lane = tid & 31;
  const int l15 = lane & 15, h = lane >> 4;
  const int bh = blockIdx.y;                   // 0..95
  const int q0 = blockIdx.x * 64;
  const int head = bh % 12;

  const unsigned short* Q  = qbuf + (size_t)bh * 1024 * 64;
  const unsigned short* Kg = kbuf + (size_t)bh * 1024 * 64;
  const unsigned short* Vg = vbuf + (size_t)bh * 1024 * 64;

  int cr[2], cc[2];
#pragma unroll
  for (int it = 0; it < 2; ++it) { int e = tid * 8 + it * 1024; cr[it] = e >> 6; cc[it] = e & 63; }

  // Q A-fragments (16 rows x 64 d) resident in registers for the whole loop.
  const int qrow = q0 + wid * 16 + l15;
  bf16x16 qa[2];
#pragma unroll
  for (int c = 0; c < 2; ++c) {
    const unsigned short* p = Q + (size_t)qrow * 64 + 32 * c;
    qa[c] = cat16(*reinterpret_cast<const bf16x8*>(p + 16 * h),
                  *reinterpret_cast<const bf16x8*>(p + 16 * h + 8));
  }

  f32x8 o[4];
#pragma unroll
  for (int i = 0; i < 4; ++i) o[i] = fzero();
  float mrow[8], lrow[8];
#pragma unroll
  for (int r = 0; r < 8; ++r) { mrow[r] = -3.0e38f; lrow[r] = 0.0f; }

  const int ntiles = q0 / 32 + 2;              // covers keys 0 .. q0+63

  // ---- prologue: stage key-tile 0 into buffer 0
#pragma unroll
  for (int it = 0; it < 2; ++it)
    async_copy_b128(lds_off(&Ks[0][cr[it] * 72 + cc[it]]),
                    Kg + (size_t)cr[it] * 64 + cc[it]);
  {
    u16x8 vreg[2];
#pragma unroll
    for (int it = 0; it < 2; ++it)
      vreg[it] = gld8(Vg + (size_t)cr[it] * 64 + cc[it]);
#pragma unroll
    for (int it = 0; it < 2; ++it)
#pragma unroll
      for (int j = 0; j < 8; ++j) Vs[0][(cc[it] + j) * 40 + cr[it]] = vreg[it][j];
  }

  for (int kt = 0; kt < ntiles; ++kt) {
    const int j0 = kt * 32;
    const int buf = kt & 1;
    async_wait0();
    __syncthreads();

    const bool more = (kt + 1) < ntiles;
    u16x8 vreg[2];
    if (more) {                                // stage key-tile kt+1 into buf^1
      const int jn = j0 + 32;
#pragma unroll
      for (int it = 0; it < 2; ++it)
        async_copy_b128(lds_off(&Ks[buf ^ 1][cr[it] * 72 + cc[it]]),
                        Kg + (size_t)(jn + cr[it]) * 64 + cc[it]);
#pragma unroll
      for (int it = 0; it < 2; ++it)
        vreg[it] = gld8(Vg + (size_t)(jn + cr[it]) * 64 + cc[it]);
    }

    // S = Q * K^T  (two 16x16 key sub-tiles, K-dim = d = 64 split in 2 WMMAs)
    f32x8 s[2];
#pragma unroll
    for (int t = 0; t < 2; ++t) {
      const int krow = t * 16 + l15;
      f32x8 z = fzero();
      bf16x16 b0 = cat16(ld8(&Ks[buf][krow * 72 + 16 * h]),
                         ld8(&Ks[buf][krow * 72 + 16 * h + 8]));
      z = WMMA_BF16(qa[0], b0, z);
      bf16x16 b1 = cat16(ld8(&Ks[buf][krow * 72 + 32 + 16 * h]),
                         ld8(&Ks[buf][krow * 72 + 32 + 16 * h + 8]));
      s[t] = WMMA_BF16(qa[1], b1, z);
    }

    // causal mask + online softmax (rows = r + 8h, cols striped over 16 lanes)
#pragma unroll
    for (int r = 0; r < 8; ++r) {
      const int qg = q0 + wid * 16 + r + 8 * h;
      float s0 = s[0][r], s1 = s[1][r];
      if (j0 + l15 > qg)      s0 = -3.0e38f;
      if (j0 + 16 + l15 > qg) s1 = -3.0e38f;
      float rm = fmaxf(s0, s1);
      rm = fmaxf(rm, __shfl_xor(rm, 1, 32));
      rm = fmaxf(rm, __shfl_xor(rm, 2, 32));
      rm = fmaxf(rm, __shfl_xor(rm, 4, 32));
      rm = fmaxf(rm, __shfl_xor(rm, 8, 32));
      const float mold = mrow[r];
      const float mnew = fmaxf(mold, rm);
      const float scale = __expf(mold - mnew);
      const float p0 = __expf(s0 - mnew);
      const float p1 = __expf(s1 - mnew);
      float psum = p0 + p1;
      psum += __shfl_xor(psum, 1, 32);
      psum += __shfl_xor(psum, 2, 32);
      psum += __shfl_xor(psum, 4, 32);
      psum += __shfl_xor(psum, 8, 32);
      mrow[r] = mnew;
      lrow[r] = lrow[r] * scale + psum;
#pragma unroll
      for (int nt = 0; nt < 4; ++nt) o[nt][r] *= scale;
      // P (C-layout) -> per-wave LDS scratch for A-layout re-read
      const int prow = wid * 16 + r + 8 * h;
      Ps[prow * 40 + l15]      = f2bf(p0);
      Ps[prow * 40 + 16 + l15] = f2bf(p1);
    }

    // O += P * V   (K-dim = 32 keys; same-wave LDS is in-order -> no barrier)
    bf16x16 pa = cat16(ld8(&Ps[(wid * 16 + l15) * 40 + 8 * h]),
                       ld8(&Ps[(wid * 16 + l15) * 40 + 16 + 8 * h]));
#pragma unroll
    for (int nt = 0; nt < 4; ++nt) {
      const int d = nt * 16 + l15;
      bf16x16 vfr = cat16(ld8(&Vs[buf][d * 40 + 8 * h]),
                          ld8(&Vs[buf][d * 40 + 16 + 8 * h]));
      o[nt] = WMMA_BF16(pa, vfr, o[nt]);
    }

    if (more) {                                // V data arrived during compute
#pragma unroll
      for (int it = 0; it < 2; ++it)
#pragma unroll
        for (int j = 0; j < 8; ++j) Vs[buf ^ 1][(cc[it] + j) * 40 + cr[it]] = vreg[it][j];
    }
  }

  // normalize and write y[B*T][768] (bf16) at column head*64 + d
#pragma unroll
  for (int r = 0; r < 8; ++r) {
    const float inv = 1.0f / lrow[r];
    const int gm = q0 + wid * 16 + r + 8 * h;
#pragma unroll
    for (int nt = 0; nt < 4; ++nt) {
      const int d = nt * 16 + l15;
      yb[(size_t)gm * 768 + head * 64 + d] = f2bf(o[nt][r] * inv);
    }
  }
}

// ---------------------------------------------------------------------------
// Output projection: [8192,768]bf16 x [768,768]bf16 + bias -> fp32 out
// block tile 128(M) x 64(N), double-buffered as in qkv_gemm.
// ---------------------------------------------------------------------------
__global__ __launch_bounds__(128) void proj_gemm(
    const unsigned short* __restrict__ yb,
    const unsigned short* __restrict__ wb,   // [768][768]
    const float* __restrict__ bias,
    float* __restrict__ out) {
  __shared__ unsigned short As[2][128 * 40];
  __shared__ unsigned short Bs[2][64 * 40];

  const int tid = threadIdx.x;
  const int wid = tid >> 5, lane = tid & 31;
  const int l15 = lane & 15, h = lane >> 4;
  const int m0 = blockIdx.y * 128;
  const int n0 = blockIdx.x * 64;

  int ar[4], ac[4];
#pragma unroll
  for (int it = 0; it < 4; ++it) { int e = tid * 8 + it * 1024; ar[it] = e >> 5; ac[it] = e & 31; }
  int br[2], bc[2];
#pragma unroll
  for (int it = 0; it < 2; ++it) { int e = tid * 8 + it * 1024; br[it] = e >> 6; bc[it] = e & 63; }

  f32x8 acc[2][4];
#pragma unroll
  for (int i = 0; i < 2; ++i)
#pragma unroll
    for (int j = 0; j < 4; ++j) acc[i][j] = fzero();

#pragma unroll
  for (int it = 0; it < 4; ++it)
    async_copy_b128(lds_off(&As[0][ar[it] * 40 + ac[it]]),
                    yb + (size_t)(m0 + ar[it]) * 768 + ac[it]);
  {
    u16x8 breg[2];
#pragma unroll
    for (int it = 0; it < 2; ++it)
      breg[it] = gld8(wb + (size_t)br[it] * 768 + n0 + bc[it]);
#pragma unroll
    for (int it = 0; it < 2; ++it)
#pragma unroll
      for (int j = 0; j < 8; ++j) Bs[0][(bc[it] + j) * 40 + br[it]] = breg[it][j];
  }

  for (int kt = 0; kt < 24; ++kt) {
    const int buf = kt & 1;
    async_wait0();
    __syncthreads();

    const bool more = (kt + 1) < 24;
    u16x8 breg[2];
    if (more) {
      const int kn = (kt + 1) * 32;
#pragma unroll
      for (int it = 0; it < 4; ++it)
        async_copy_b128(lds_off(&As[buf ^ 1][ar[it] * 40 + ac[it]]),
                        yb + (size_t)(m0 + ar[it]) * 768 + kn + ac[it]);
#pragma unroll
      for (int it = 0; it < 2; ++it)
        breg[it] = gld8(wb + (size_t)(kn + br[it]) * 768 + n0 + bc[it]);
    }

    bf16x16 a[2];
#pragma unroll
    for (int ms = 0; ms < 2; ++ms) {
      const int am = wid * 32 + ms * 16 + l15;
      a[ms] = cat16(ld8(&As[buf][am * 40 + 8 * h]),
                    ld8(&As[buf][am * 40 + 16 + 8 * h]));
    }
#pragma unroll
    for (int nt = 0; nt < 4; ++nt) {
      const int bn = nt * 16 + l15;
      bf16x16 b = cat16(ld8(&Bs[buf][bn * 40 + 8 * h]),
                        ld8(&Bs[buf][bn * 40 + 16 + 8 * h]));
#pragma unroll
      for (int ms = 0; ms < 2; ++ms) acc[ms][nt] = WMMA_BF16(a[ms], b, acc[ms][nt]);
    }

    if (more) {
#pragma unroll
      for (int it = 0; it < 2; ++it)
#pragma unroll
        for (int j = 0; j < 8; ++j) Bs[buf ^ 1][(bc[it] + j) * 40 + br[it]] = breg[it][j];
    }
  }

#pragma unroll
  for (int ms = 0; ms < 2; ++ms) {
#pragma unroll
    for (int nt = 0; nt < 4; ++nt) {
#pragma unroll
      for (int r = 0; r < 8; ++r) {
        int gm = m0 + wid * 32 + ms * 16 + r + 8 * h;
        int gn = n0 + nt * 16 + l15;
        out[(size_t)gm * 768 + gn] = acc[ms][nt][r] + bias[gn];
      }
    }
  }
}

// ---------------------------------------------------------------------------
extern "C" void kernel_launch(void* const* d_in, const int* in_sizes, int n_in,
                              void* d_out, int out_size, void* d_ws, size_t ws_size,
                              hipStream_t stream) {
  (void)in_sizes; (void)n_in; (void)out_size; (void)ws_size;
  const float* x      = (const float*)d_in[0];
  const float* w_attn = (const float*)d_in[1];
  const float* b_attn = (const float*)d_in[2];
  const float* w_proj = (const float*)d_in[3];
  const float* b_proj = (const float*)d_in[4];
  float* out = (float*)d_out;

  char* ws = (char*)d_ws;
  size_t off = 0;
  auto take = [&](size_t bytes) -> char* {
    char* p = ws + off;
    off += (bytes + 255) & ~(size_t)255;
    return p;
  };
  unsigned short* xb  = (unsigned short*)take((size_t)8192 * 768 * 2);
  unsigned short* wab = (unsigned short*)take((size_t)768 * 2304 * 2);
  unsigned short* wpb = (unsigned short*)take((size_t)768 * 768 * 2);
  unsigned short* qb  = (unsigned short*)take((size_t)96 * 1024 * 64 * 2);
  unsigned short* kb  = (unsigned short*)take((size_t)96 * 1024 * 64 * 2);
  unsigned short* vb  = (unsigned short*)take((size_t)96 * 1024 * 64 * 2);
  unsigned short* ybuf = (unsigned short*)take((size_t)8192 * 768 * 2);

  cvt_f32_bf16<<<dim3(6144), 256, 0, stream>>>(x, xb, 8192 * 768);
  cvt_f32_bf16<<<dim3(1728), 256, 0, stream>>>(w_attn, wab, 768 * 2304);
  cvt_f32_bf16<<<dim3(576),  256, 0, stream>>>(w_proj, wpb, 768 * 768);

  qkv_gemm<<<dim3(36, 64), 128, 0, stream>>>(xb, wab, b_attn, qb, kb, vb);
  attn_kernel<<<dim3(16, 96), 128, 0, stream>>>(qb, kb, vb, ybuf);
  proj_gemm<<<dim3(12, 64), 128, 0, stream>>>(ybuf, wpb, b_proj, out);
}
